// PerformerBlock_2190433321796
// MI455X (gfx1250) — matrix operand order repaired and checked by reading, not compiled, not run on hardware
//
#include <hip/hip_runtime.h>

#define BATCH 4
#define SEQ   4096
#define MDIM  512
#define HEADS 8
#define DH    64
#define NBH   32           // BATCH*HEADS
#define MF    266          // nb_features
#define MP    288          // padded to 9*32 for exact K-chunking
#define EPSK  1e-4f
#define NORMALIZER 0.35355339059327379f   // 64^-0.25
#define RATIO      0.06131393394849658f   // 266^-0.5

typedef __attribute__((ext_vector_type(16))) __bf16 v16bf;
typedef __attribute__((ext_vector_type(2)))  __bf16 v2bf;
typedef __attribute__((ext_vector_type(8)))  float  v8f;

__device__ __forceinline__ unsigned short f32_to_bf16_bits(float f) {
  unsigned u = __float_as_uint(f);
  unsigned r = u + 0x7FFFu + ((u >> 16) & 1u);   // round-to-nearest-even
  return (unsigned short)(r >> 16);
}
#if __has_builtin(__builtin_amdgcn_cvt_pk_bf16_f32)
__device__ __forceinline__ unsigned pack_bf16x2(float lo, float hi) {
  v2bf r = __builtin_amdgcn_cvt_pk_bf16_f32(lo, hi);   // single v_cvt_pk_bf16_f32
  return __builtin_bit_cast(unsigned, r);
}
#else
__device__ __forceinline__ unsigned pack_bf16x2(float lo, float hi) {
  return (unsigned)f32_to_bf16_bits(lo) | ((unsigned)f32_to_bf16_bits(hi) << 16);
}
#endif
__device__ __forceinline__ float bf16_bits_to_f32(unsigned short h) {
  return __uint_as_float(((unsigned)h) << 16);
}
__device__ __forceinline__ __bf16 bits2bf(unsigned short h) {
  return __builtin_bit_cast(__bf16, h);
}
__device__ __forceinline__ v8f wmma_bf16(v16bf a, v16bf b, v8f c) {
  return __builtin_amdgcn_wmma_f32_16x16x32_bf16(false, a, false, b, (short)0, c, false, false);
}

// LDS byte offset of a __shared__ pointer (generic LDS addr carries offset in low 32 bits).
__device__ __forceinline__ unsigned lds_off(const void* p) {
  return (unsigned)(unsigned long long)p;
}
// gfx1250 async global->LDS DMA, 16B per active lane, tracked by ASYNCcnt.
__device__ __forceinline__ void async_b128(unsigned ldsByteOff, const void* gaddr) {
  asm volatile("global_load_async_to_lds_b128 %0, %1, off"
               :: "v"(ldsByteOff), "v"(gaddr) : "memory");
}
__device__ __forceinline__ void wait_async0() {
  asm volatile("s_wait_asynccnt 0" ::: "memory");
}

// A fragment 16x32 bf16 from row-major [M][K] LDS tile: one contiguous 32B run/lane.
__device__ __forceinline__ v16bf load_frag_a(const unsigned short* p0, int ld, int lane) {
  int half = lane >> 4, row = lane & 15;
  const unsigned short* p = p0 + row * ld + half * 16;
  v16bf a;
#pragma unroll
  for (int j = 0; j < 8; ++j) a[j] = bits2bf(p[j]);          // K = half*16 + j
#pragma unroll
  for (int j = 0; j < 8; ++j) a[8 + j] = bits2bf(p[8 + j]);  // K = half*16 + 8 + j
  return a;
}
// B fragment 32x16 from N-major [N][K] tile: one contiguous 32B run per lane.
__device__ __forceinline__ v16bf load_frag_bT(const unsigned short* p0, int ldk, int lane) {
  int half = lane >> 4, col = lane & 15;
  const unsigned short* p = p0 + col * ldk + half * 16;
  v16bf b;
#pragma unroll
  for (int j = 0; j < 16; ++j) b[j] = bits2bf(p[j]);
  return b;
}

// ---------------------------------------------------------------- utilities
__global__ void zero_f32_kernel(float* p, int n) {
  int i = blockIdx.x * 256 + threadIdx.x;
  if (i < n) p[i] = 0.f;
}

__global__ void projcvt_kernel(const float* __restrict__ proj, unsigned short* __restrict__ projbf) {
  int i = blockIdx.x * 256 + threadIdx.x;
  if (i >= MP * DH) return;
  int m = i / DH, d = i - m * DH;
  float v = (m < MF) ? proj[m * DH + d] : 0.f;
  projbf[i] = f32_to_bf16_bits(v);
}

// ---------------------------------------------------- 512-K GEMM (QKV + out-proj)
// C(16384x512) = A(16384x512) @ W^T.  128x64 WG tile, 32x32 register tile per wave
// (2 A-frags x 2 B-frags -> 4 WMMAs per 32-K chunk).
// MODE 0: scatter to (b,h,n,DH).  MODE 1: +bias+resid -> (b,n,DIM).  MODE 2: vT (b,h,DH,n).
template <int MODE>
__global__ void gemm512_kernel(const float* __restrict__ A, const float* __restrict__ W,
                               const float* __restrict__ bias, const float* __restrict__ resid,
                               float* __restrict__ out) {
  __shared__ unsigned short As[128][32];   // [m][k] 8 KB
  __shared__ unsigned short Bt[64][32];    // [n][k] 4 KB
  int tid = threadIdx.x;
  int wave = tid >> 5, lane = tid & 31;
  int rowBase = blockIdx.x * 128;
  int colBase = blockIdx.y * 64;
  int wRow = (wave >> 1) * 32;   // 4 row groups of 32
  int wCol = (wave & 1) * 32;    // 2 col groups of 32
  int ar = tid >> 1, as = (tid & 1) * 16;   // A staging: 16 contiguous k / thread
  int bc = tid >> 2, bs = (tid & 3) * 8;    // B staging: 8 contiguous k / thread
  const float* arow = A + (size_t)(rowBase + ar) * MDIM + as;
  const float* brow = W + (size_t)(colBase + bc) * MDIM + bs;
  v8f acc00 = {}, acc01 = {}, acc10 = {}, acc11 = {};
  for (int kb = 0; kb < MDIM; kb += 32) {
    __syncthreads();
    if (kb + 32 < MDIM) {                     // global_prefetch_b8 next chunk
      __builtin_prefetch(arow + kb + 32, 0, 1);
      __builtin_prefetch(brow + kb + 32, 0, 1);
    }
    {
      const float4* pa = (const float4*)(arow + kb);
      float4 x0 = pa[0], x1 = pa[1], x2 = pa[2], x3 = pa[3];
      uint4 w0, w1;
      w0.x = pack_bf16x2(x0.x, x0.y); w0.y = pack_bf16x2(x0.z, x0.w);
      w0.z = pack_bf16x2(x1.x, x1.y); w0.w = pack_bf16x2(x1.z, x1.w);
      w1.x = pack_bf16x2(x2.x, x2.y); w1.y = pack_bf16x2(x2.z, x2.w);
      w1.z = pack_bf16x2(x3.x, x3.y); w1.w = pack_bf16x2(x3.z, x3.w);
      *(uint4*)&As[ar][as] = w0;
      *(uint4*)&As[ar][as + 8] = w1;
      const float4* pb = (const float4*)(brow + kb);
      float4 y0 = pb[0], y1 = pb[1];
      uint4 wb;
      wb.x = pack_bf16x2(y0.x, y0.y); wb.y = pack_bf16x2(y0.z, y0.w);
      wb.z = pack_bf16x2(y1.x, y1.y); wb.w = pack_bf16x2(y1.z, y1.w);
      *(uint4*)&Bt[bc][bs] = wb;
    }
    __syncthreads();
    v16bf a0 = load_frag_a(&As[wRow][0], 32, lane);
    v16bf a1 = load_frag_a(&As[wRow + 16][0], 32, lane);
    v16bf b0 = load_frag_bT(&Bt[wCol][0], 32, lane);
    v16bf b1 = load_frag_bT(&Bt[wCol + 16][0], 32, lane);
    acc00 = wmma_bf16(a0, b0, acc00);
    acc01 = wmma_bf16(a0, b1, acc01);
    acc10 = wmma_bf16(a1, b0, acc10);
    acc11 = wmma_bf16(a1, b1, acc11);
  }
  int half = lane >> 4, colL = lane & 15;
  auto store_tile = [&](v8f acc, int ri, int ci) {
    int col = colBase + wCol + ci * 16 + colL;
#pragma unroll
    for (int r = 0; r < 8; ++r) {
      int row = rowBase + wRow + ri * 16 + half * 8 + r;
      float v = acc[r];
      if (MODE == 0) {
        int bb = row >> 12, nn = row & 4095;
        int h = col >> 6, dd = col & 63;
        out[(((size_t)(bb * HEADS + h)) * SEQ + nn) * DH + dd] = v;
      } else if (MODE == 2) {
        int bb = row >> 12, nn = row & 4095;
        int h = col >> 6, dd = col & 63;
        out[(((size_t)(bb * HEADS + h)) * DH + dd) * SEQ + nn] = v;  // contiguous in r
      } else {
        out[(size_t)row * MDIM + col] = v + bias[col] + resid[(size_t)row * MDIM + col];
      }
    }
  };
  store_tile(acc00, 0, 0);
  store_tile(acc01, 0, 1);
  store_tile(acc10, 1, 0);
  store_tile(acc11, 1, 1);
}

// --------------------------------------------- softmax-kernel feature map
// MODE 0: query (per-row max) -> qp [bh][n][MP]
// MODE 1: key pass1 (per-WG partial of global max)
// MODE 2: key pass2 -> kpT [bh][MP][n] (feature-major), accumulate k_sum
template <int MODE>
__global__ void feature_kernel(const float* __restrict__ T,               // [NBH][SEQ][DH]
                               const unsigned short* __restrict__ projbf, // [MP][DH] bf16
                               unsigned short* __restrict__ outP,
                               float* __restrict__ kmax_part,             // [NBH][SEQ/32]
                               const float* __restrict__ kmax,            // [NBH]
                               float* __restrict__ ksum) {                // [NBH][MP]
  __shared__ unsigned short As[32][64];  // scaled input tile (bf16)
  __shared__ float td[32][MP];
  __shared__ float diag[32];
  __shared__ float rmax[32];
  __shared__ float red[256];
  int tid = threadIdx.x;
  int wave = tid >> 5, lane = tid & 31;
  int bh = blockIdx.x;
  int rowBase = blockIdx.y * 32;
  const float* Tp = T + ((size_t)bh * SEQ + rowBase) * DH;

  if (tid < 32) diag[tid] = 0.f;
  __syncthreads();
  {
    int ar = tid >> 3, as = (tid & 7) * 8;   // 8 contiguous d / thread
    const float4* pa = (const float4*)(Tp + (size_t)ar * DH + as);
    float4 x0 = pa[0], x1 = pa[1];
    float ss = x0.x * x0.x + x0.y * x0.y + x0.z * x0.z + x0.w * x0.w +
               x1.x * x1.x + x1.y * x1.y + x1.z * x1.z + x1.w * x1.w;
    atomicAdd(&diag[ar], ss);
    uint4 w;
    w.x = pack_bf16x2(x0.x * NORMALIZER, x0.y * NORMALIZER);
    w.y = pack_bf16x2(x0.z * NORMALIZER, x0.w * NORMALIZER);
    w.z = pack_bf16x2(x1.x * NORMALIZER, x1.y * NORMALIZER);
    w.w = pack_bf16x2(x1.z * NORMALIZER, x1.w * NORMALIZER);
    *(uint4*)&As[ar][as] = w;
  }
  __syncthreads();

  // 2 row-groups x 18 col-tiles, K=64 in two 32-chunks; B straight from projbf
  for (int t = wave; t < 2 * 18; t += 8) {
    int rg = t / 18, ct = t % 18;
    v8f acc = {};
#pragma unroll
    for (int kc = 0; kc < 2; ++kc) {
      v16bf a = load_frag_a(&As[rg * 16][kc * 32], 64, lane);
      v16bf b = load_frag_bT(projbf + (size_t)(ct * 16) * DH + kc * 32, DH, lane);
      acc = wmma_bf16(a, b, acc);
    }
    int half = lane >> 4, colL = lane & 15;
#pragma unroll
    for (int r = 0; r < 8; ++r) td[rg * 16 + half * 8 + r][ct * 16 + colL] = acc[r];
  }
  __syncthreads();

  if (MODE == 0) {
    if (tid < 32) {
      float mx = -1e30f;
      for (int m = 0; m < MF; ++m) mx = fmaxf(mx, td[tid][m]);
      rmax[tid] = mx;
    }
    __syncthreads();
    unsigned short* op = outP + ((size_t)bh * SEQ + rowBase) * MP;
    for (int r = 0; r < 32; ++r) {
      float dterm = 0.0625f * diag[r] + rmax[r];   // 0.5*normalizer^2 = 0.0625
      for (int m = tid; m < MP; m += 256) {
        float val = (m < MF) ? RATIO * (__expf(td[r][m] - dterm) + EPSK) : 0.f;
        op[(size_t)r * MP + m] = f32_to_bf16_bits(val);
      }
    }
  } else if (MODE == 1) {
    float mx = -1e30f;
    for (int r = 0; r < 32; ++r)
      for (int m = tid; m < MF; m += 256) mx = fmaxf(mx, td[r][m]);
    red[tid] = mx;
    __syncthreads();
    for (int s = 128; s > 0; s >>= 1) {
      if (tid < s) red[tid] = fmaxf(red[tid], red[tid + s]);
      __syncthreads();
    }
    if (tid == 0) kmax_part[bh * (SEQ / 32) + blockIdx.y] = red[0];
  } else {
    float gmx = kmax[bh];
    unsigned short* op = outP + (size_t)bh * MP * SEQ;   // feature-major
    for (int m = tid; m < MP; m += 256) {
      float s = 0.f;
      unsigned short* dst = op + (size_t)m * SEQ + rowBase;  // 32 contiguous bf16
#pragma unroll
      for (int r = 0; r < 32; ++r) {
        float val = (m < MF) ? RATIO * (__expf(td[r][m] - 0.0625f * diag[r] - gmx) + EPSK) : 0.f;
        dst[r] = f32_to_bf16_bits(val);
        s += val;
      }
      if (m < MF) atomicAdd(&ksum[bh * MP + m], s);
    }
  }
}

__global__ void kmax_reduce_kernel(const float* __restrict__ part, float* __restrict__ kmax) {
  __shared__ float red[128];
  int bh = blockIdx.x, t = threadIdx.x;
  red[t] = part[bh * 128 + t];
  __syncthreads();
  for (int s = 64; s > 0; s >>= 1) {
    if (t < s) red[t] = fmaxf(red[t], red[t + s]);
    __syncthreads();
  }
  if (t == 0) kmax[bh] = red[0];
}

// ------------------------- context^T = (kp^T @ v)^T per head, K = SEQ
// A[m][k=n] = kpT (row-major, async DMA to LDS); B^T[d][k=n] = vT (contiguous).
__global__ void context_kernel(const unsigned short* __restrict__ kpT, // [NBH][MP][SEQ] bf16
                               const float* __restrict__ vT,           // [NBH][DH][SEQ]
                               float* __restrict__ ctxT) {             // [NBH][DH][MP]
  __shared__ unsigned short As[16][32];   // A[m][k]
  __shared__ unsigned short Bt[64][32];   // B^T[d][k]
  int tid = threadIdx.x;
  int wave = tid >> 5, lane = tid & 31;
  int bh = blockIdx.x, mBase = blockIdx.y * 16;
  const unsigned short* kpp = kpT + ((size_t)bh * MP + mBase) * SEQ;
  const float* vp = vT + (size_t)bh * DH * SEQ;
  int am = tid >> 2, asg = (tid & 3) * 8;   // A: threads 0..63, 16B async each
  int bd = tid >> 1, bsg = (tid & 1) * 16;  // B: 16 contiguous k / thread
  unsigned aoff = lds_off(&As[am & 15][asg]);
  v8f acc = {};
  for (int kb = 0; kb < SEQ; kb += 32) {
    __syncthreads();
    if (tid < 64) async_b128(aoff, kpp + (size_t)am * SEQ + kb + asg);
    {
      const float4* pb = (const float4*)(vp + (size_t)bd * SEQ + kb + bsg);
      float4 x0 = pb[0], x1 = pb[1], x2 = pb[2], x3 = pb[3];
      uint4 w0, w1;
      w0.x = pack_bf16x2(x0.x, x0.y); w0.y = pack_bf16x2(x0.z, x0.w);
      w0.z = pack_bf16x2(x1.x, x1.y); w0.w = pack_bf16x2(x1.z, x1.w);
      w1.x = pack_bf16x2(x2.x, x2.y); w1.y = pack_bf16x2(x2.z, x2.w);
      w1.z = pack_bf16x2(x3.x, x3.y); w1.w = pack_bf16x2(x3.z, x3.w);
      *(uint4*)&Bt[bd][bsg] = w0;
      *(uint4*)&Bt[bd][bsg + 8] = w1;
    }
    wait_async0();
    __syncthreads();
    v16bf a = load_frag_a(&As[0][0], 32, lane);
    v16bf b = load_frag_bT(&Bt[wave * 16][0], 32, lane);
    acc = wmma_bf16(a, b, acc);
  }
  int half = lane >> 4, colL = lane & 15;
  float* cout = ctxT + ((size_t)bh * DH + wave * 16 + colL) * MP + mBase + half * 8;
#pragma unroll
  for (int r = 0; r < 8; ++r) cout[r] = acc[r];
}

// -------------------------------------- d_inv[n] = 1 / (qp[n,:] . k_sum)
__global__ void dinv_kernel(const unsigned short* __restrict__ qp, const float* __restrict__ ksum,
                            float* __restrict__ dinv) {
  __shared__ float ks[MP];
  int bh = blockIdx.x;
  int tid = threadIdx.x;
  for (int m = tid; m < MP; m += 256) ks[m] = ksum[bh * MP + m];
  __syncthreads();
  int n = blockIdx.y * 256 + tid;
  const unsigned short* qr = qp + ((size_t)bh * SEQ + n) * MP;
  float s = 0.f;
  for (int m = 0; m < MF; ++m) s += bf16_bits_to_f32(qr[m]) * ks[m];
  dinv[bh * SEQ + n] = 1.0f / s;
}

// -------------------------------------- out = (qp @ ctx) * d_inv -> (b,n,DIM)
// A = qp rows (bf16, async DMA); B^T[d][m] = ctxT rows (contiguous).
__global__ void outgemm_kernel(const unsigned short* __restrict__ qp,  // [NBH][SEQ][MP] bf16
                               const float* __restrict__ ctxT,         // [NBH][DH][MP]
                               const float* __restrict__ dinv,         // [NBH][SEQ]
                               float* __restrict__ out_attn) {         // [BATCH][SEQ][MDIM]
  __shared__ unsigned short As[32][MP];   // [n][m]  18 KB
  __shared__ unsigned short Bt[64][MP];   // [d][m]  36 KB
  int tid = threadIdx.x;
  int wave = tid >> 5, lane = tid & 31;
  int bh = blockIdx.x;
  int rowBase = blockIdx.y * 32;
  const unsigned short* qpp = qp + ((size_t)bh * SEQ + rowBase) * MP;
  const float* cp = ctxT + (size_t)bh * DH * MP;
  const int SEGS = MP / 8;  // 36
  for (int i = tid; i < 32 * SEGS; i += 256) {          // async A: 1152 x 16B
    int r = i / SEGS, sg = i - r * SEGS;
    async_b128(lds_off(&As[r][sg * 8]), qpp + (size_t)r * MP + sg * 8);
  }
  for (int i = tid; i < 64 * SEGS; i += 256) {          // B: cvt f32->bf16
    int d = i / SEGS, sg = i - d * SEGS;
    const float4* pb = (const float4*)(cp + (size_t)d * MP + sg * 8);
    float4 x0 = pb[0], x1 = pb[1];
    uint4 w;
    w.x = pack_bf16x2(x0.x, x0.y); w.y = pack_bf16x2(x0.z, x0.w);
    w.z = pack_bf16x2(x1.x, x1.y); w.w = pack_bf16x2(x1.z, x1.w);
    *(uint4*)&Bt[d][sg * 8] = w;
  }
  wait_async0();
  __syncthreads();
  int wRow = (wave >> 2) * 16;  // 2 row groups
  int wCol = (wave & 3) * 16;   // 4 col groups
  v8f acc = {};
#pragma unroll
  for (int kb = 0; kb < MP; kb += 32) {
    v16bf a = load_frag_a(&As[wRow][kb], MP, lane);
    v16bf b = load_frag_bT(&Bt[wCol][kb], MP, lane);
    acc = wmma_bf16(a, b, acc);
  }
  int half = lane >> 4, colL = lane & 15;
  int bb = bh >> 3, h = bh & 7;
#pragma unroll
  for (int r = 0; r < 8; ++r) {
    int row = rowBase + wRow + half * 8 + r;
    float val = acc[r] * dinv[bh * SEQ + row];
    out_attn[((size_t)bb * SEQ + row) * MDIM + h * DH + wCol + colL] = val;
  }
}

// ---------------------------------------------------------------- launcher
extern "C" void kernel_launch(void* const* d_in, const int* in_sizes, int n_in,
                              void* d_out, int out_size, void* d_ws, size_t ws_size,
                              hipStream_t stream) {
  (void)in_sizes; (void)n_in; (void)out_size; (void)ws_size;
  const float* x    = (const float*)d_in[0];
  const float* Wq   = (const float*)d_in[1];
  const float* Wk   = (const float*)d_in[2];
  const float* Wv   = (const float*)d_in[3];
  const float* Wo   = (const float*)d_in[4];
  const float* bo   = (const float*)d_in[5];
  const float* proj = (const float*)d_in[6];
  float* y = (float*)d_out;

  char* p = (char*)d_ws;
  auto alloc = [&](size_t bytes) {
    char* r = p;
    p += (bytes + 255) & ~(size_t)255;
    return r;
  };
  float*          q      = (float*)alloc((size_t)NBH * SEQ * DH * 4);
  float*          k      = (float*)alloc((size_t)NBH * SEQ * DH * 4);
  float*          vT     = (float*)alloc((size_t)NBH * DH * SEQ * 4);
  unsigned short* qp     = (unsigned short*)alloc((size_t)NBH * SEQ * MP * 2);
  unsigned short* kpT    = (unsigned short*)alloc((size_t)NBH * MP * SEQ * 2);
  unsigned short* projbf = (unsigned short*)alloc((size_t)MP * DH * 2);
  float*          ctxT   = (float*)alloc((size_t)NBH * DH * MP * 4);
  float*          ksum   = (float*)alloc((size_t)NBH * MP * 4);
  float*          kmaxp  = (float*)alloc((size_t)NBH * 128 * 4);
  float*          kmax   = (float*)alloc((size_t)NBH * 4);
  float*          dinv   = (float*)alloc((size_t)NBH * SEQ * 4);
  float*          oat    = (float*)alloc((size_t)BATCH * SEQ * MDIM * 4);

  zero_f32_kernel<<<dim3((NBH * MP + 255) / 256), 256, 0, stream>>>(ksum, NBH * MP);
  projcvt_kernel<<<dim3((MP * DH + 255) / 256), 256, 0, stream>>>(proj, projbf);

  dim3 g512(BATCH * SEQ / 128, MDIM / 64);
  gemm512_kernel<0><<<g512, 256, 0, stream>>>(x, Wq, nullptr, nullptr, q);
  gemm512_kernel<0><<<g512, 256, 0, stream>>>(x, Wk, nullptr, nullptr, k);
  gemm512_kernel<2><<<g512, 256, 0, stream>>>(x, Wv, nullptr, nullptr, vT);

  dim3 gf(NBH, SEQ / 32);
  feature_kernel<0><<<gf, 256, 0, stream>>>(q, projbf, qp, nullptr, nullptr, nullptr);
  feature_kernel<1><<<gf, 256, 0, stream>>>(k, projbf, nullptr, kmaxp, nullptr, nullptr);
  kmax_reduce_kernel<<<NBH, 128, 0, stream>>>(kmaxp, kmax);
  feature_kernel<2><<<gf, 256, 0, stream>>>(k, projbf, kpT, nullptr, kmax, ksum);

  context_kernel<<<dim3(NBH, MP / 16), 128, 0, stream>>>(kpT, vT, ctxT);
  dinv_kernel<<<dim3(NBH, SEQ / 256), 256, 0, stream>>>(qp, ksum, dinv);
  outgemm_kernel<<<dim3(NBH, SEQ / 32), 256, 0, stream>>>(qp, ctxT, dinv, oat);

  gemm512_kernel<1><<<g512, 256, 0, stream>>>(oat, Wo, bo, x, y);
}
